// GRUAmp_24240795419118
// MI455X (gfx1250) — matrix operand order repaired and checked by reading, not compiled, not run on hardware
//
#include <hip/hip_runtime.h>
#include <hip/hip_bf16.h>

// GRU scan for MI455X (gfx1250, wave32, WMMA).
// Grid = 2 workgroups x 96 threads (3 waves). Each WG owns 16 batch rows.
// Wave 0 = r gate, wave 1 = z gate, wave 2 = n gate + state update.
// Recurrent matmul per step: [16,64] x [64,192] via v_wmma_f32_16x16x32_f16.

typedef __attribute__((ext_vector_type(16))) _Float16 v16h;
typedef __attribute__((ext_vector_type(8)))  float    v8f;

#define B_TOT 32
#define T_LEN 32768
#define H_DIM 64
#define NB    16   // batch rows per workgroup

union HU { v16h v; _Float16 h[16]; uint4 q[2]; };
union FU { v8f  v; float    f[8]; };

#define LOG2E 1.44269504088896340736f

// Fast transcendentals: single v_rcp_f32 / v_exp_f32 / v_tanh_f32 instead of
// the IEEE div expansion (v_div_scale + Newton + v_div_fmas) clang emits for '/'.
__device__ __forceinline__ float fast_rcp(float x)  { return __builtin_amdgcn_rcpf(x); }
__device__ __forceinline__ float fast_exp2(float x) { return __builtin_amdgcn_exp2f(x); }

#if __has_builtin(__builtin_amdgcn_tanhf)
__device__ __forceinline__ float tanh_fast(float x) {
  return __builtin_amdgcn_tanhf(x);                      // hw V_TANH_F32
}
__device__ __forceinline__ float sigmoid_fast(float x) {
  return fmaf(0.5f, __builtin_amdgcn_tanhf(0.5f * x), 0.5f);
}
#else
__device__ __forceinline__ float tanh_fast(float x) {
  float e = fast_exp2(2.0f * LOG2E * x);                 // exp(2x)
  return 1.0f - 2.0f * fast_rcp(e + 1.0f);
}
__device__ __forceinline__ float sigmoid_fast(float x) {
  return fast_rcp(1.0f + fast_exp2(-LOG2E * x));
}
#endif

__global__ __launch_bounds__(96, 1)
void gru_scan_wmma(const float* __restrict__ x,
                   const float* __restrict__ w_ih,
                   const float* __restrict__ w_hh,
                   const float* __restrict__ b_ih,
                   const float* __restrict__ b_hh,
                   const float* __restrict__ w_lin,
                   const float* __restrict__ b_lin,
                   float* __restrict__ out)
{
  __shared__ __align__(16) _Float16 hbuf[NB * H_DIM];  // h_t as f16, [m][j]
  __shared__ __align__(16) float    rbuf[NB * H_DIM];  // sigmoid(r) per step
  __shared__ __align__(16) float    zbuf[NB * H_DIM];  // sigmoid(z) per step
  __shared__ __align__(16) float    xbuf[2][NB];       // double-buffered x_t

  const int tid  = threadIdx.x;
  const int wave = tid >> 5;      // 0=r, 1=z, 2=n
  const int lane = tid & 31;
  const int lhal = lane & 15;
  const int lhi  = lane >> 4;     // 0 for lanes 0-15, 1 for lanes 16-31
  const int b0   = blockIdx.x * NB;

  // ---- Pack this wave's slice of w_hh^T into WMMA B-layout registers.
  // B (32x16 f16 per tile): lanes 0-15 hold K 0-7 and 16-23,
  // lanes 16-31 hold K 8-15 and 24-31; column N = lane&15.
  v16h Bt[4][2];
  #pragma unroll
  for (int ni = 0; ni < 4; ++ni) {
    const int n = wave * H_DIM + ni * 16 + lhal;      // w_hh row (gate output)
    #pragma unroll
    for (int ki = 0; ki < 2; ++ki) {
      const int kbase = ki * 32 + lhi * 8;
      HU u;
      #pragma unroll
      for (int i = 0; i < 16; ++i)
        u.h[i] = (_Float16)w_hh[n * H_DIM + kbase + i + (i >= 8 ? 8 : 0)];
      Bt[ni][ki] = u.v;
    }
  }

  // Per-lane gate constants (match D-layout column j = ni*16 + (lane&15)).
  float wih_l[4], bsum_l[4], bih_l[4], bhh_l[4];
  #pragma unroll
  for (int ni = 0; ni < 4; ++ni) {
    const int n = wave * H_DIM + ni * 16 + lhal;
    wih_l[ni]  = w_ih[n];
    bih_l[ni]  = b_ih[n];
    bhh_l[ni]  = b_hh[n];
    bsum_l[ni] = b_ih[n] + b_hh[n];
  }

  // Per-lane w_lin constants matching the A-register half-index mapping:
  // half i of tile ki holds h[m][ ki*32 + lhi*8 + i + (i>=8 ? 8 : 0) ].
  float wl[2][16];
  #pragma unroll
  for (int ki = 0; ki < 2; ++ki)
    #pragma unroll
    for (int i = 0; i < 16; ++i)
      wl[ki][i] = w_lin[ki * 32 + lhi * 8 + i + (i >= 8 ? 8 : 0)];
  const float blin = b_lin[0];

  // h_0 = 0: A registers and n-wave's D-layout state start at zero.
  v16h A[2]    = {};
  v8f  hold[4] = {};

  // Prologue: stage x_0 into LDS.
  if (wave == 0 && lane < NB)
    xbuf[0][lane] = x[(b0 + lane) * T_LEN];
  __syncthreads();

  for (int t = 0; t < T_LEN; ++t) {
    const int cur = t & 1, nxt = cur ^ 1;

    // 1) Recurrent projection: acc[ni] = h_{t-1} @ w_hh^T (K=64 -> 2 chained).
    v8f acc[4];
    #pragma unroll
    for (int ni = 0; ni < 4; ++ni) {
      v8f c = {};
      c = __builtin_amdgcn_wmma_f32_16x16x32_f16(false, A[0], false, Bt[ni][0],
                                                 (short)0, c, false, false);
      c = __builtin_amdgcn_wmma_f32_16x16x32_f16(false, A[1], false, Bt[ni][1],
                                                 (short)0, c, false, false);
      acc[ni] = c;
    }

    // x_t in D-layout: element v of a D tile sits at row m = v + 8*lhi.
    float4 xa  = *reinterpret_cast<const float4*>(&xbuf[cur][lhi * 8]);
    float4 xbv = *reinterpret_cast<const float4*>(&xbuf[cur][lhi * 8 + 4]);
    float xm[8] = {xa.x, xa.y, xa.z, xa.w, xbv.x, xbv.y, xbv.z, xbv.w};

    // Prefetch x_{t+1} (one contiguous stream per lane -> cache friendly).
    float xnext = 0.0f;
    if (wave == 0 && lane < NB && (t + 1) < T_LEN)
      xnext = x[(b0 + lane) * T_LEN + t + 1];

    // 2) r/z gates: sigmoid, publish to LDS in [m][j] order.
    if (wave < 2) {
      float* gbuf = (wave == 0) ? rbuf : zbuf;
      #pragma unroll
      for (int ni = 0; ni < 4; ++ni) {
        FU c; c.v = acc[ni];
        #pragma unroll
        for (int v = 0; v < 8; ++v) {
          const int m = v + lhi * 8;
          const int j = ni * 16 + lhal;
          float arg = c.f[v] + fmaf(xm[v], wih_l[ni], bsum_l[ni]);
          gbuf[m * H_DIM + j] = sigmoid_fast(arg);
        }
      }
    }
    __syncthreads();   // barrier 1: r/z visible to n wave

    // 3) n gate + state update (n wave keeps h in D-layout registers).
    if (wave == 2) {
      #pragma unroll
      for (int ni = 0; ni < 4; ++ni) {
        FU c; c.v = acc[ni];
        FU h; h.v = hold[ni];
        #pragma unroll
        for (int v = 0; v < 8; ++v) {
          const int m = v + lhi * 8;
          const int j = ni * 16 + lhal;
          float r  = rbuf[m * H_DIM + j];
          float z  = zbuf[m * H_DIM + j];
          float xn = fmaf(xm[v], wih_l[ni], bih_l[ni]);
          float nn = tanh_fast(xn + r * (c.f[v] + bhh_l[ni]));
          float hn = nn + z * (h.f[v] - nn);   // (1-z)*n + z*h
          h.f[v] = hn;
          hbuf[m * H_DIM + j] = (_Float16)hn;
        }
        hold[ni] = h.v;
      }
    } else if (wave == 0 && lane < NB) {
      xbuf[nxt][lane] = xnext;
    }
    __syncthreads();   // barrier 2: h_t (f16) + x_{t+1} visible to all

    // 4) Repack h_t into WMMA A-layout (16-bit A 16x32):
    //    lanes 0-15 row m: K 0-7 in v0-3, K 16-23 in v4-7; lanes 16-31: +8.
    #pragma unroll
    for (int ki = 0; ki < 2; ++ki) {
      const int jb = ki * 32 + lhi * 8;
      HU u;
      u.q[0] = *reinterpret_cast<const uint4*>(&hbuf[lhal * H_DIM + jb]);
      u.q[1] = *reinterpret_cast<const uint4*>(&hbuf[lhal * H_DIM + jb + 16]);
      A[ki] = u.v;
    }

    // 5) Readout y_t = x_t + h_t . w_lin + b_lin, straight from A registers.
    if (wave == 0) {
      float part = 0.0f;
      #pragma unroll
      for (int ki = 0; ki < 2; ++ki) {
        HU u; u.v = A[ki];
        #pragma unroll
        for (int i = 0; i < 16; ++i)
          part = fmaf((float)u.h[i], wl[ki][i], part);
      }
      part += __shfl_xor(part, 16, 32);  // combine K-halves of each row
      if (lane < NB)
        out[(b0 + lane) * T_LEN + t] = xbuf[cur][lane] + part + blin;
    }
  }

  // Final state [1,B,H] appended after y.
  if (wave == 2) {
    float* st = out + (size_t)B_TOT * T_LEN;
    #pragma unroll
    for (int ni = 0; ni < 4; ++ni) {
      FU h; h.v = hold[ni];
      #pragma unroll
      for (int v = 0; v < 8; ++v) {
        const int m = v + lhi * 8;
        const int j = ni * 16 + lhal;
        st[(b0 + m) * H_DIM + j] = h.f[v];
      }
    }
  }
}

extern "C" void kernel_launch(void* const* d_in, const int* in_sizes, int n_in,
                              void* d_out, int out_size, void* d_ws, size_t ws_size,
                              hipStream_t stream) {
  const float* x     = (const float*)d_in[0];
  const float* w_ih  = (const float*)d_in[1];
  const float* w_hh  = (const float*)d_in[2];
  const float* b_ih  = (const float*)d_in[3];
  const float* b_hh  = (const float*)d_in[4];
  const float* w_lin = (const float*)d_in[5];
  const float* b_lin = (const float*)d_in[6];
  gru_scan_wmma<<<dim3(2), dim3(96), 0, stream>>>(
      x, w_ih, w_hh, b_ih, b_hh, w_lin, b_lin, (float*)d_out);
}
